// CoreAttention_74852690035361
// MI455X (gfx1250) — matrix operand order repaired and checked by reading, not compiled, not run on hardware
//
#include <hip/hip_runtime.h>

// ---------------------------------------------------------------------------
// Types for gfx1250 WMMA (wave32): v16bf A/B fragments, v8f C/D accumulators.
// ---------------------------------------------------------------------------
typedef __attribute__((ext_vector_type(16))) __bf16 v16bf;
typedef __attribute__((ext_vector_type(8)))  __bf16 v8bf;
typedef __attribute__((ext_vector_type(8)))  float  v8f;

#define S_SEQ 2048
#define NHEAD 16
#define HDIM  128
#define SCALE 0.088388347648318447f   // 1/sqrt(2048/16) ; COEFF == 1

static __device__ inline v8f vzero8() {
  v8f z;
#pragma unroll
  for (int i = 0; i < 8; ++i) z[i] = 0.0f;
  return z;
}

static __device__ inline v8f wmma_bf16(v16bf a, v16bf b, v8f c) {
  // (neg_a, A, neg_b, B, c_mod, C, reuse_a, reuse_b)
  return __builtin_amdgcn_wmma_f32_16x16x32_bf16(false, a, false, b,
                                                 (short)0, c, false, false);
}

// B fragment (32x16): lane holds col n=L&15; element i -> K = i + (L>=16?16:0).
// Caller passes pointer to the 16 contiguous bf16 this lane needs.
static __device__ inline v16bf load_b_contig(const __bf16* p) {
  v8bf lo = *(const v8bf*)(p);
  v8bf hi = *(const v8bf*)(p + 8);
  return __builtin_shufflevector(lo, hi, 0,1,2,3,4,5,6,7,8,9,10,11,12,13,14,15);
}

// A fragment (16x32, row-major source): lane holds row m=L&15.
// Element i -> K = hl*16 + i.
static __device__ inline v16bf load_a_rm(const __bf16* row, int hl) {
  v8bf lo = *(const v8bf*)(row + hl * 16);
  v8bf hi = *(const v8bf*)(row + hl * 16 + 8);
  return __builtin_shufflevector(lo, hi, 0,1,2,3,4,5,6,7,8,9,10,11,12,13,14,15);
}

// Same A layout but converting from fp32 source.
static __device__ inline v16bf load_a_f32(const float* row, int hl) {
  v16bf a;
#pragma unroll
  for (int i = 0; i < 8; ++i) a[i] = (__bf16)row[hl * 16 + i];
#pragma unroll
  for (int i = 0; i < 8; ++i) a[8 + i] = (__bf16)row[hl * 16 + 8 + i];
  return a;
}

// XOR butterfly within the wave via ds_swizzle (masks <16 stay inside a
// 16-lane half, which is exactly the lane group sharing one score row set).
template <int M>
static __device__ inline float xshuf(float x) {
  int xi = __builtin_bit_cast(int, x);
  xi = __builtin_amdgcn_ds_swizzle(xi, 0x1f | (M << 10));
  return __builtin_bit_cast(float, xi);
}
static __device__ inline float half_max(float v) {
  v = fmaxf(v, xshuf<1>(v)); v = fmaxf(v, xshuf<2>(v));
  v = fmaxf(v, xshuf<4>(v)); v = fmaxf(v, xshuf<8>(v));
  return v;
}
static __device__ inline float half_sum(float v) {
  v += xshuf<1>(v); v += xshuf<2>(v); v += xshuf<4>(v); v += xshuf<8>(v);
  return v;
}

// Async global -> LDS copy of 16B per lane (ASYNCcnt tracked, CDNA5 path).
static __device__ inline void async_copy16(unsigned lds_off, const void* gptr) {
  asm volatile("global_load_async_to_lds_b128 %0, %1, off"
               :: "v"(lds_off), "v"(gptr)
               : "memory");
}
static __device__ inline void wait_async0() {
  asm volatile("s_wait_asynccnt 0x0" ::: "memory");
}

// ---------------------------------------------------------------------------
// Stage 0: svd[h][d][e] (f32) -> svdT[h][e][d] (bf16) so WMMA B-fragments of
// the projection GEMM are contiguous 32B per lane.
// ---------------------------------------------------------------------------
__global__ void svdT_kernel(const float* __restrict__ svd_qk,
                            const float* __restrict__ svd_v,
                            __bf16* __restrict__ qkT,
                            __bf16* __restrict__ vT) {
  int idx   = blockIdx.x * 256 + threadIdx.x;   // 2 * 16 * 128 * 128 total
  int which = idx >> 18;
  int r     = idx & ((1 << 18) - 1);
  int h = r >> 14, e = (r >> 7) & 127, d = r & 127;
  const float* src = which ? svd_v : svd_qk;
  __bf16*      dst = which ? vT    : qkT;
  dst[((size_t)(h * 128 + e)) * 128 + d] =
      (__bf16)src[((size_t)(h * 128 + d)) * 128 + e];
}

// ---------------------------------------------------------------------------
// Stage 1: per-head projection, one wave per 16x128 tile.
//   mat 0: qp[h][s][e] = Q[s,0,h,:] @ svd_qk[h]   (row-major bf16)
//   mat 1: kp[h][s][e] = K ...                    (row-major bf16)
//   mat 2: vpT[h][e][s] = (V @ svd_v[h])^T        (transposed bf16)
// ---------------------------------------------------------------------------
__global__ __launch_bounds__(128)
void proj_kernel(const float* __restrict__ Qin, const float* __restrict__ Kin,
                 const float* __restrict__ Vin,
                 const __bf16* __restrict__ svdqkT,
                 const __bf16* __restrict__ svdvT,
                 __bf16* __restrict__ qp, __bf16* __restrict__ kp,
                 __bf16* __restrict__ vpT) {
  const int lane = threadIdx.x & 31;
  const int wave = threadIdx.x >> 5;
  const int tile = blockIdx.x * 4 + wave;          // 0 .. 3*16*128-1
  if (tile >= 3 * NHEAD * (S_SEQ / 16)) return;
  const int mat   = tile / (NHEAD * (S_SEQ / 16));
  const int rem   = tile % (NHEAD * (S_SEQ / 16));
  const int h     = rem / (S_SEQ / 16);
  const int sbase = (rem % (S_SEQ / 16)) * 16;
  const int ln = lane & 15, hl = lane >> 4;

  const float*  in = (mat == 0) ? Qin : (mat == 1) ? Kin : Vin;
  const __bf16* bT = (mat == 2) ? svdvT : svdqkT;

  // A fragments: rows of the raw input [S,1,H,D]
  const float* arow = in + ((size_t)(sbase + ln) * NHEAD + h) * HDIM;
  v16bf a[4];
#pragma unroll
  for (int kc = 0; kc < 4; ++kc) a[kc] = load_a_f32(arow + kc * 32, hl);

#pragma unroll
  for (int nt = 0; nt < 8; ++nt) {
    v8f acc = vzero8();
#pragma unroll
    for (int kc = 0; kc < 4; ++kc) {
      const __bf16* brow =
          bT + ((size_t)(h * HDIM + nt * 16 + ln)) * HDIM + kc * 32 + hl * 16;
      acc = wmma_bf16(a[kc], load_b_contig(brow), acc);
    }
    if (mat < 2) {
      __bf16* outp = ((mat == 0) ? qp : kp) +
                     ((size_t)h * S_SEQ + sbase) * HDIM;
#pragma unroll
      for (int r = 0; r < 8; ++r)
        outp[(size_t)(r + hl * 8) * HDIM + nt * 16 + ln] = (__bf16)acc[r];
    } else {
      v8bf pk;
#pragma unroll
      for (int r = 0; r < 8; ++r) pk[r] = (__bf16)acc[r];
      *(v8bf*)(vpT + ((size_t)(h * HDIM + nt * 16 + ln)) * S_SEQ + sbase +
               hl * 8) = pk;
    }
  }
}

// ---------------------------------------------------------------------------
// Stage 2: flash attention. One workgroup = 4 waves = 4 consecutive q-tiles
// of one head, sharing async-staged K/V key blocks in LDS (double-buffered).
// ---------------------------------------------------------------------------
__global__ __launch_bounds__(128)
void attn_kernel(const __bf16* __restrict__ qp, const __bf16* __restrict__ kp,
                 const __bf16* __restrict__ vpT, float* __restrict__ out) {
  __shared__ __bf16 Klds[2][32 * HDIM];   // [buf][krow][e]   8KB each
  __shared__ __bf16 Vlds[2][HDIM * 32];   // [buf][e][krow]   8KB each
  __shared__ __bf16 Plds[4][16 * 32];     // per-wave P staging

  const int lane = threadIdx.x & 31;
  const int wave = threadIdx.x >> 5;
  const int h     = blockIdx.x >> 5;            // 16 heads
  const int qblk  = blockIdx.x & 31;            // 32 blocks of 64 q rows
  const int qbase = qblk * 64 + wave * 16;      // this wave's 16 q rows
  const int ln = lane & 15, hl = lane >> 4;

  // Q A-fragments (16 rows x 128 K)
  const __bf16* qrow = qp + ((size_t)h * S_SEQ + qbase + ln) * HDIM;
  v16bf aq[4];
#pragma unroll
  for (int kc = 0; kc < 4; ++kc) aq[kc] = load_a_rm(qrow + kc * 32, hl);

  v8f acc[8];
#pragma unroll
  for (int nt = 0; nt < 8; ++nt) acc[nt] = vzero8();
  float mrow[8], lrow[8];
#pragma unroll
  for (int r = 0; r < 8; ++r) { mrow[r] = -1e30f; lrow[r] = 0.0f; }

  __bf16* P = &Plds[wave][0];
  const int nkb = 2 * qblk + 2;   // causal block count for the widest wave

  // --- cooperative async stage of one 32-key block into LDS buffer `buf` ---
  auto stage = [&](int kb, int buf) {
    const int kbase = kb * 32;
    // K tile: rows [kbase, kbase+32) of kp[h] are 8KB contiguous.
    // Wave copies its contiguous 2KB quarter (4 x 512B instructions).
    const __bf16* gk = kp + ((size_t)h * S_SEQ + kbase) * HDIM + wave * 1024;
    unsigned lk = (unsigned)(uintptr_t)&Klds[buf][wave * 1024];
#pragma unroll
    for (int j = 0; j < 4; ++j)
      async_copy16(lk + j * 512 + lane * 16, gk + j * 256 + lane * 8);
    // V tile: 32-col slice of vpT rows; wave covers e in [wave*32, wave*32+32).
#pragma unroll
    for (int j = 0; j < 4; ++j) {
      const int e = wave * 32 + j * 8 + (lane >> 2);
      const __bf16* gv =
          vpT + ((size_t)(h * HDIM + e)) * S_SEQ + kbase + (lane & 3) * 8;
      unsigned lv = (unsigned)(uintptr_t)&Vlds[buf][e * 32 + (lane & 3) * 8];
      async_copy16(lv, gv);
    }
  };

  stage(0, 0);
  for (int kb = 0; kb < nkb; ++kb) {
    const int cur = kb & 1;
    wait_async0();        // my copies for block kb are done
    __syncthreads();      // => everyone's copies are done; prev buf free
    if (kb + 1 < nkb) stage(kb + 1, 1 - cur);   // overlaps with compute below

    const int kbase = kb * 32;
    if (kbase < qbase + 16) {   // causal: this wave still has live columns
      const __bf16* Kl = &Klds[cur][0];
      const __bf16* Vl = &Vlds[cur][0];

      // ---- S = Q K^T for key cols [kbase, kbase+32) ---------------------
      v8f s0 = vzero8(), s1 = vzero8();
      const __bf16* krow0 = Kl + (size_t)ln * HDIM;
      const __bf16* krow1 = Kl + (size_t)(16 + ln) * HDIM;
#pragma unroll
      for (int kc = 0; kc < 4; ++kc) {
        s0 = wmma_bf16(aq[kc], load_b_contig(krow0 + kc * 32 + hl * 16), s0);
        s1 = wmma_bf16(aq[kc], load_b_contig(krow1 + kc * 32 + hl * 16), s1);
      }

      // ---- causal mask + online softmax ---------------------------------
      const int c0 = kbase + ln, c1 = kbase + 16 + ln;
#pragma unroll
      for (int r = 0; r < 8; ++r) {
        const int q = qbase + r + hl * 8;
        float v0 = (c0 <= q) ? s0[r] * SCALE : -10000.0f;
        float v1 = (c1 <= q) ? s1[r] * SCALE : -10000.0f;
        float mx = half_max(fmaxf(v0, v1));
        float mnew = fmaxf(mrow[r], mx);
        float p0 = __expf(v0 - mnew);
        float p1 = __expf(v1 - mnew);
        float sum = half_sum(p0 + p1);
        float alpha = __expf(mrow[r] - mnew);
        lrow[r] = lrow[r] * alpha + sum;
        mrow[r] = mnew;
#pragma unroll
        for (int nt = 0; nt < 8; ++nt) acc[nt][r] *= alpha;
        P[(r + hl * 8) * 32 + ln]      = (__bf16)p0;
        P[(r + hl * 8) * 32 + 16 + ln] = (__bf16)p1;
      }
      // Same-wave LDS RAW: DS ops are in-order per wave; wait + clobber keeps
      // the compiler from hoisting the A-fragment loads over the stores.
      asm volatile("s_wait_dscnt 0x0" ::: "memory");

      // ---- acc += P V  (A = P 16x32 from LDS, B = V tile from LDS) ------
      v16bf ap = load_a_rm(P + ln * 32, hl);
#pragma unroll
      for (int nt = 0; nt < 8; ++nt) {
        acc[nt] =
            wmma_bf16(ap, load_b_contig(Vl + (nt * 16 + ln) * 32 + hl * 16),
                      acc[nt]);
      }
    }
  }

  // ---- epilogue: out[q][h*128+e] = acc / l -------------------------------
#pragma unroll
  for (int r = 0; r < 8; ++r) {
    const int q = qbase + r + hl * 8;
    const float inv = 1.0f / lrow[r];
#pragma unroll
    for (int nt = 0; nt < 8; ++nt)
      out[(size_t)q * (NHEAD * HDIM) + h * HDIM + nt * 16 + ln] =
          acc[nt][r] * inv;
  }
}

// ---------------------------------------------------------------------------
extern "C" void kernel_launch(void* const* d_in, const int* in_sizes, int n_in,
                              void* d_out, int out_size, void* d_ws,
                              size_t ws_size, hipStream_t stream) {
  const float* Q      = (const float*)d_in[0];
  const float* K      = (const float*)d_in[1];
  const float* V      = (const float*)d_in[2];
  // d_in[3] = attention_mask (causal; computed analytically)
  const float* svd_qk = (const float*)d_in[4];
  const float* svd_v  = (const float*)d_in[5];

  char* ws = (char*)d_ws;
  __bf16* svdqkT = (__bf16*)(ws);                          // 512 KB
  __bf16* svdvT  = (__bf16*)(ws + (512u << 10));           // 512 KB
  __bf16* qp     = (__bf16*)(ws + (1u  << 20));            // 8 MB
  __bf16* kp     = (__bf16*)(ws + (9u  << 20));            // 8 MB
  __bf16* vpT    = (__bf16*)(ws + (17u << 20));            // 8 MB
  float*  out    = (float*)d_out;

  svdT_kernel<<<2048, 256, 0, stream>>>(svd_qk, svd_v, svdqkT, svdvT);
  proj_kernel<<<(3 * NHEAD * (S_SEQ / 16)) / 4, 128, 0, stream>>>(
      Q, K, V, svdqkT, svdvT, qp, kp, vpT);
  attn_kernel<<<NHEAD * 32, 128, 0, stream>>>(qp, kp, vpT, out);
}
